// ImprovedTransformerBlock_38835094291129
// MI455X (gfx1250) — matrix operand order repaired and checked by reading, not compiled, not run on hardware
//
#include <hip/hip_runtime.h>
#include <hip/hip_bf16.h>
#include <math.h>

// ---------------------------------------------------------------------------
// Types for CDNA5 WMMA (wave32). 16x16x32 bf16 -> f32 accumulate.
// ---------------------------------------------------------------------------
typedef __attribute__((ext_vector_type(16))) __bf16 v16bf;
typedef __attribute__((ext_vector_type(8)))  float  v8f;

union Frag16 {            // 16 bf16 = 32 bytes; filled with two 16B LDS loads
    v16bf v;
    uint4 q[2];
};

__device__ __forceinline__ unsigned short f2bf(float f) {
    unsigned int u = __float_as_uint(f);
    unsigned int r = u + 0x7FFFu + ((u >> 16) & 1u);   // round-to-nearest-even
    return (unsigned short)(r >> 16);
}
__device__ __forceinline__ float bf2f(unsigned short h) {
    return __uint_as_float(((unsigned int)h) << 16);
}
__device__ __forceinline__ float gelu_exact(float v) {
    return 0.5f * v * (1.0f + erff(v * 0.70710678118654752440f));
}

static constexpr int NTOK = 8192;      // B*L
static constexpr int CDIM = 1024;
static constexpr int FDIM = 4096;
static constexpr int NEXP = 8;
static constexpr int SEQL = 4096;      // per-batch length (== 1<<12)

// ---------------------------------------------------------------------------
// Generic bf16 WMMA GEMM:  out = epilogue( A[M,K](bf16) x B[K,N](f32->bf16) )
//   BM=128, BN=128, BK=32 ; 256 threads = 8 waves ; wave owns 32x64 (2x4 tiles)
//   MODE 0: plain rows          (M, N multiples of 128 -> no guards at all)
//   MODE 1: row shift by +/-2, zero-filled outside the owning 4096-token batch
//   MODE 2: MoE gather/scatter rows with device-side count
//   Double-buffered LDS: global loads for tile k+1 are staged in registers
//   while the 8 WMMAs of tile k issue; one s_barrier per K tile.
// ---------------------------------------------------------------------------
template <int MODE>
__global__ __launch_bounds__(256) void gemm_bf16_kernel(
    const unsigned short* __restrict__ A, int lda,
    const float* __restrict__ Bw, long long strideK, long long strideN,
    const float* __restrict__ bias,
    float* __restrict__ outF, const float* __restrict__ resid,
    unsigned short* __restrict__ outB,
    int M, int N, int K, int doGelu, int shift,
    const int* __restrict__ gatherA, const int* __restrict__ scatterOut,
    const int* __restrict__ cntPtr)
{
    constexpr int BM = 128, BN = 128, BK = 32;

    int cnt = M;
    if (MODE == 2) {
        cnt = *cntPtr;                               // device-side row count
        if ((int)(blockIdx.x * BM) >= cnt) return;   // uniform early-exit
    }
    const int mBase = blockIdx.x * BM;
    const int nBase = blockIdx.y * BN;

    __shared__ unsigned short sA[2][BM][BK + 8];     // row stride 80B (16B-aligned)
    __shared__ unsigned short sBt[2][BN][BK + 8];    // B stored K-transposed

    const int tid  = threadIdx.x;
    const int lane = tid & 31, wave = tid >> 5;
    const int wm = wave >> 1, wn = wave & 1;         // 4x2 wave grid
    const int hl = lane >> 4, l15 = lane & 15;

    // ---- K-invariant staging addresses, resolved ONCE (no loop divergence) --
    const int rA  = tid >> 1;                        // A stage: 2 threads/row
    const int c0  = (tid & 1) * 16;
    const unsigned short* aPtr;                      // null => zero-fill row
    {
        const int m = mBase + rA;
        if (MODE == 2) {
            aPtr = (m < cnt) ? (A + (long long)gatherA[m] * lda + c0) : nullptr;
        } else if (MODE == 1) {
            const int s2 = m + shift;
            const bool ok = (s2 >= 0) && (s2 < M) && ((s2 >> 12) == (m >> 12));
            aPtr = ok ? (A + (long long)s2 * lda + c0) : nullptr;
        } else {
            aPtr = A + (long long)m * lda + c0;
        }
    }
    const int kB = tid >> 3;                         // B stage: fixed k per thread
    const int n0 = (tid & 7) * 16;                   // 16 consecutive n
    const float* bPtr = Bw + (long long)kB * strideK + (long long)(nBase + n0) * strideN;

    v8f acc[2][4];
    #pragma unroll
    for (int i = 0; i < 2; ++i)
        #pragma unroll
        for (int j = 0; j < 4; ++j) acc[i][j] = v8f{};

    const int nk = K / BK;

    // ---- prologue: stage tile 0 into buffer 0 ----
    {
        uint4 a0, a1;
        if (MODE == 0 || aPtr) {
            const uint4* p = (const uint4*)aPtr;
            a0 = p[0]; a1 = p[1];
        } else {
            a0 = uint4{0u,0u,0u,0u}; a1 = a0;
        }
        *(uint4*)&sA[0][rA][c0]     = a0;
        *(uint4*)&sA[0][rA][c0 + 8] = a1;
        #pragma unroll
        for (int t = 0; t < 16; ++t)
            sBt[0][n0 + t][kB] = f2bf(bPtr[(long long)t * strideN]);
    }
    __syncthreads();

    for (int kt = 0; kt < nk; ++kt) {
        const int buf  = kt & 1;
        const bool more = (kt + 1 < nk);

        // ---- issue global loads for tile kt+1 (registers) ----
        uint4 a0, a1;
        float bv[16];
        if (more) {
            if (MODE == 0 || aPtr) {
                const uint4* p = (const uint4*)(aPtr + (kt + 1) * BK);
                a0 = p[0]; a1 = p[1];
                if (kt + 2 < nk)   // speculative prefetch (global_prefetch_b8)
                    __builtin_prefetch((const void*)(aPtr + (kt + 2) * BK), 0, 0);
            } else {
                a0 = uint4{0u,0u,0u,0u}; a1 = a0;
            }
            const float* bp = bPtr + (long long)(kt + 1) * BK * strideK;
            #pragma unroll
            for (int t = 0; t < 16; ++t) bv[t] = bp[(long long)t * strideN];
        }

        // ---- compute tile kt: fragments per ISA 16-bit A 16x32 / B 32x16 ----
        Frag16 afr[2], bfr[4];
        #pragma unroll
        for (int i = 0; i < 2; ++i) {
            const int row = wm * 32 + i * 16 + l15;
            afr[i].q[0] = *(const uint4*)&sA[buf][row][8 * hl];      // K = 8*hl..+7
            afr[i].q[1] = *(const uint4*)&sA[buf][row][16 + 8 * hl]; // K = 16+8*hl..+7
        }
        #pragma unroll
        for (int j = 0; j < 4; ++j) {
            const int col = wn * 64 + j * 16 + l15;
            bfr[j].q[0] = *(const uint4*)&sBt[buf][col][8 * hl];      // K = 8*hl..+7
            bfr[j].q[1] = *(const uint4*)&sBt[buf][col][16 + 8 * hl]; // K = 16+8*hl..+7
        }
        #pragma unroll
        for (int i = 0; i < 2; ++i)
            #pragma unroll
            for (int j = 0; j < 4; ++j)
                acc[i][j] = __builtin_amdgcn_wmma_f32_16x16x32_bf16(
                    false, afr[i].v, false, bfr[j].v, (short)0, acc[i][j], false, false);

        // ---- drain staged registers into the other LDS buffer ----
        if (more) {
            const int nb = buf ^ 1;
            *(uint4*)&sA[nb][rA][c0]     = a0;
            *(uint4*)&sA[nb][rA][c0 + 8] = a1;
            #pragma unroll
            for (int t = 0; t < 16; ++t) sBt[nb][n0 + t][kB] = f2bf(bv[t]);
        }
        __syncthreads();
    }

    // ---- epilogue: C/D layout -> M = r + 8*(lane>>4), N = lane&15 ----
    #pragma unroll
    for (int i = 0; i < 2; ++i) {
        #pragma unroll
        for (int j = 0; j < 4; ++j) {
            #pragma unroll
            for (int r = 0; r < 8; ++r) {
                const int m = mBase + wm * 32 + i * 16 + r + 8 * hl;
                const int n = nBase + wn * 64 + j * 16 + l15;
                if (MODE != 2 || m < cnt) {
                    float v = acc[i][j][r];
                    if (bias)   v += bias[n];
                    if (doGelu) v = gelu_exact(v);
                    if (outF) {
                        const int om = (MODE == 2) ? scatterOut[m] : m;
                        const float rr = resid ? resid[(long long)m * N + n] : 0.0f;
                        outF[(long long)om * N + n] = rr + v;
                    }
                    if (outB) outB[(long long)m * N + n] = f2bf(v);
                }
            }
        }
    }
}

// ---------------------------------------------------------------------------
// LayerNorm over C=1024, one block per token, bf16 output (feeds WMMA GEMMs)
// ---------------------------------------------------------------------------
__global__ __launch_bounds__(256) void ln_kernel(
    const float* __restrict__ x, const float* __restrict__ g,
    const float* __restrict__ b, unsigned short* __restrict__ out)
{
    const int t = blockIdx.x;
    const float* xr = x + (long long)t * CDIM;
    __shared__ float red[256];

    float s = 0.0f, s2 = 0.0f;
    for (int c = threadIdx.x; c < CDIM; c += 256) {
        const float v = xr[c];
        s += v; s2 += v * v;
    }
    red[threadIdx.x] = s; __syncthreads();
    for (int o = 128; o > 0; o >>= 1) {
        if (threadIdx.x < o) red[threadIdx.x] += red[threadIdx.x + o];
        __syncthreads();
    }
    const float mean = red[0] * (1.0f / CDIM); __syncthreads();
    red[threadIdx.x] = s2; __syncthreads();
    for (int o = 128; o > 0; o >>= 1) {
        if (threadIdx.x < o) red[threadIdx.x] += red[threadIdx.x + o];
        __syncthreads();
    }
    const float var = red[0] * (1.0f / CDIM) - mean * mean;
    const float rs  = rsqrtf(var + 1e-5f);
    for (int c = threadIdx.x; c < CDIM; c += 256)
        out[(long long)t * CDIM + c] = f2bf((xr[c] - mean) * rs * g[c] + b[c]);
}

// ---------------------------------------------------------------------------
// Window attention, faithful to the reference's reshape semantics.
// Only windows 0..15 per (b,h) survive the [:L] slice. Per window:
//   S = Q(64xK256) Kt  -> WMMA over 4 s-chunks ; softmax rows (e axis)
//   O = P(64x64) V(64x256) -> WMMA per s-chunk ; scatter-store with
//   p = n*256 + d*4 + s/64 , c = h*64 + s%64
// ---------------------------------------------------------------------------
__global__ __launch_bounds__(256) void attn_kernel(
    const unsigned short* __restrict__ qkv, unsigned short* __restrict__ outO)
{
    const int blk = blockIdx.x;
    const int n = blk & 15;
    const int h = (blk >> 4) & 15;
    const int b = blk >> 8;
    const int base = b * SEQL + n * 128;
    const int ldq = 3 * CDIM;

    __shared__ unsigned short sQ[64][72];   // [d][s-chunk]
    __shared__ unsigned short sK[64][72];   // [e][s-chunk]  (B K-transposed)
    __shared__ unsigned short sV[64][72];   // [s][e]        (B K-transposed)
    __shared__ float          sS[64][64];
    __shared__ unsigned short sP[64][72];

    const int tid = threadIdx.x;
    const int lane = tid & 31, wave = tid >> 5;
    const int wm = wave >> 1, wn = wave & 1;
    const int hl = lane >> 4, l15 = lane & 15;

    v8f accS[2] = {v8f{}, v8f{}};

    // ---------- S = Q K^T, accumulated over 4 chunks of s ----------
    for (int sc = 0; sc < 4; ++sc) {
        const int sl = tid >> 2;
        const int d0 = (tid & 3) * 16;
        const long long row = (long long)(base + sc * 64 + sl) * ldq + h * 64 + d0;
        union { uint4 q[2]; unsigned short u[16]; } uq, uk;
        uq.q[0] = ((const uint4*)(qkv + row))[0];
        uq.q[1] = ((const uint4*)(qkv + row))[1];
        uk.q[0] = ((const uint4*)(qkv + row + CDIM))[0];
        uk.q[1] = ((const uint4*)(qkv + row + CDIM))[1];
        #pragma unroll
        for (int t = 0; t < 16; ++t) { sQ[d0 + t][sl] = uq.u[t]; sK[d0 + t][sl] = uk.u[t]; }
        __syncthreads();

        #pragma unroll
        for (int kk = 0; kk < 2; ++kk) {
            const int kb = kk * 32;
            Frag16 a, bf[2];
            const int rd = wm * 16 + l15;
            a.q[0] = *(const uint4*)&sQ[rd][kb + 8 * hl];
            a.q[1] = *(const uint4*)&sQ[rd][kb + 16 + 8 * hl];
            #pragma unroll
            for (int j = 0; j < 2; ++j) {
                const int e = wn * 32 + j * 16 + l15;
                bf[j].q[0] = *(const uint4*)&sK[e][kb + 8 * hl];
                bf[j].q[1] = *(const uint4*)&sK[e][kb + 16 + 8 * hl];
            }
            #pragma unroll
            for (int j = 0; j < 2; ++j)
                accS[j] = __builtin_amdgcn_wmma_f32_16x16x32_bf16(
                    false, a.v, false, bf[j].v, (short)0, accS[j], false, false);
        }
        __syncthreads();
    }

    // scale by hd^-0.5 = 1/8 and park in LDS
    #pragma unroll
    for (int j = 0; j < 2; ++j)
        #pragma unroll
        for (int r = 0; r < 8; ++r) {
            const int d = wm * 16 + r + 8 * hl;
            const int e = wn * 32 + j * 16 + l15;
            sS[d][e] = accS[j][r] * 0.125f;
        }
    __syncthreads();

    // row softmax over e (64 wide)
    if (tid < 64) {
        float mx = -3.4e38f;
        for (int e = 0; e < 64; ++e) mx = fmaxf(mx, sS[tid][e]);
        float sum = 0.0f;
        for (int e = 0; e < 64; ++e) { const float v = expf(sS[tid][e] - mx); sS[tid][e] = v; sum += v; }
        const float inv = 1.0f / sum;
        for (int e = 0; e < 64; ++e) sP[tid][e] = f2bf(sS[tid][e] * inv);
    }
    __syncthreads();

    // ---------- O = P V, per s-chunk ----------
    for (int sc = 0; sc < 4; ++sc) {
        const int sl = tid >> 2;
        const int e0 = (tid & 3) * 16;
        const long long row = (long long)(base + sc * 64 + sl) * ldq + 2 * CDIM + h * 64 + e0;
        *(uint4*)&sV[sl][e0]     = ((const uint4*)(qkv + row))[0];
        *(uint4*)&sV[sl][e0 + 8] = ((const uint4*)(qkv + row))[1];
        __syncthreads();

        v8f accO[2] = {v8f{}, v8f{}};
        #pragma unroll
        for (int kk = 0; kk < 2; ++kk) {
            const int kb = kk * 32;
            Frag16 a, bf[2];
            const int rd = wm * 16 + l15;
            a.q[0] = *(const uint4*)&sP[rd][kb + 8 * hl];
            a.q[1] = *(const uint4*)&sP[rd][kb + 16 + 8 * hl];
            #pragma unroll
            for (int j = 0; j < 2; ++j) {
                const int s2 = wn * 32 + j * 16 + l15;
                bf[j].q[0] = *(const uint4*)&sV[s2][kb + 8 * hl];
                bf[j].q[1] = *(const uint4*)&sV[s2][kb + 16 + 8 * hl];
            }
            #pragma unroll
            for (int j = 0; j < 2; ++j)
                accO[j] = __builtin_amdgcn_wmma_f32_16x16x32_bf16(
                    false, a.v, false, bf[j].v, (short)0, accO[j], false, false);
        }
        #pragma unroll
        for (int j = 0; j < 2; ++j)
            #pragma unroll
            for (int r = 0; r < 8; ++r) {
                const int d = wm * 16 + r + 8 * hl;
                const int s = sc * 64 + wn * 32 + j * 16 + l15;
                const int p = n * 256 + d * 4 + (s >> 6);
                const int c = h * 64 + (s & 63);
                outO[(long long)(b * SEQL + p) * CDIM + c] = f2bf(accO[j][r]);
            }
        __syncthreads();
    }
}

// ---------------------------------------------------------------------------
// MoE gating: one wave per token; entropy buffer (deterministic reduce later),
// int-atomic per-expert counts + compacted token lists (order-independent).
// ---------------------------------------------------------------------------
__global__ __launch_bounds__(256) void gate_kernel(
    const unsigned short* __restrict__ xbf, const float* __restrict__ gW,
    const float* __restrict__ gb, float* __restrict__ topkP,
    float* __restrict__ entBuf, int* __restrict__ idxA,
    int* __restrict__ idxOut, int* __restrict__ ctr)
{
    const int wave = threadIdx.x >> 5, lane = threadIdx.x & 31;
    const int t = blockIdx.x * 8 + wave;
    if (t >= NTOK) return;

    float acc[NEXP];
    #pragma unroll
    for (int e = 0; e < NEXP; ++e) acc[e] = 0.0f;

    for (int c = lane; c < CDIM; c += 32) {
        const float xv = bf2f(xbf[(long long)t * CDIM + c]);
        const float* w = gW + c * NEXP;
        #pragma unroll
        for (int e = 0; e < NEXP; ++e) acc[e] += xv * w[e];
    }
    #pragma unroll
    for (int e = 0; e < NEXP; ++e)
        for (int o = 16; o > 0; o >>= 1) acc[e] += __shfl_xor(acc[e], o);

    if (lane == 0) {
        float p[NEXP], mx = -3.4e38f;
        #pragma unroll
        for (int e = 0; e < NEXP; ++e) { p[e] = acc[e] + gb[e]; mx = fmaxf(mx, p[e]); }
        float sum = 0.0f;
        #pragma unroll
        for (int e = 0; e < NEXP; ++e) { p[e] = expf(p[e] - mx); sum += p[e]; }
        const float inv = 1.0f / sum;
        float ent = 0.0f;
        #pragma unroll
        for (int e = 0; e < NEXP; ++e) { p[e] *= inv; ent -= p[e] * logf(p[e] + 1e-10f); }
        entBuf[t] = ent;

        int i0 = 0;
        #pragma unroll
        for (int e = 1; e < NEXP; ++e) if (p[e] > p[i0]) i0 = e;
        int i1 = (i0 == 0) ? 1 : 0;
        #pragma unroll
        for (int e = 0; e < NEXP; ++e) if (e != i0 && p[e] > p[i1]) i1 = e;
        const int sel[2] = {i0, i1};
        #pragma unroll
        for (int k = 0; k < 2; ++k) {
            const int e = sel[k];
            const int slot = atomicAdd(&ctr[e], 1);
            idxA[e * NTOK + slot]   = t;          // gather source row
            idxOut[e * NTOK + slot] = t * 2 + k;  // scatter (token,slot) row
            topkP[t * 2 + k] = p[e];
        }
    }
}

__global__ __launch_bounds__(256) void combine_kernel(
    float* __restrict__ xb, const float* __restrict__ moe,
    const float* __restrict__ topkP, unsigned short* __restrict__ xbf)
{
    const int t = blockIdx.x;
    const float p0 = topkP[t * 2], p1 = topkP[t * 2 + 1];
    for (int c = threadIdx.x; c < CDIM; c += 256) {
        const long long i = (long long)t * CDIM + c;
        const float v = xb[i] + p0 * moe[(long long)(t * 2) * CDIM + c]
                              + p1 * moe[(long long)(t * 2 + 1) * CDIM + c];
        xb[i] = v;
        xbf[i] = f2bf(v);
    }
}

__global__ __launch_bounds__(256) void aux_kernel(
    const float* __restrict__ entBuf, const int* __restrict__ ctr,
    float* __restrict__ outAux)
{
    __shared__ float red[256];
    float s = 0.0f;
    for (int i = threadIdx.x; i < NTOK; i += 256) s += entBuf[i];
    red[threadIdx.x] = s; __syncthreads();
    for (int o = 128; o > 0; o >>= 1) {
        if (threadIdx.x < o) red[threadIdx.x] += red[threadIdx.x + o];
        __syncthreads();
    }
    if (threadIdx.x == 0) {
        float pen = 0.0f;
        for (int e = 0; e < NEXP; ++e) {
            const float f = (float)ctr[e] / (float)NTOK - 0.3f;
            if (f > 0.0f) pen += f;
        }
        *outAux = 0.1f * (red[0] / (float)NTOK) + pen;
    }
}

__global__ void init_kernel(int* __restrict__ ctr)
{
    if (threadIdx.x < NEXP) ctr[threadIdx.x] = 0;
}

// ---------------------------------------------------------------------------
// Host side
// ---------------------------------------------------------------------------
extern "C" void kernel_launch(void* const* d_in, const int* in_sizes, int n_in,
                              void* d_out, int out_size, void* d_ws, size_t ws_size,
                              hipStream_t stream) {
    (void)in_sizes; (void)n_in; (void)out_size; (void)ws_size;
    const float* x_in   = (const float*)d_in[0];
    const float* g1     = (const float*)d_in[1];
    const float* b1     = (const float*)d_in[2];
    const float* g2     = (const float*)d_in[3];
    const float* b2     = (const float*)d_in[4];
    const float* g3     = (const float*)d_in[5];
    const float* b3     = (const float*)d_in[6];
    const float* Wqkv   = (const float*)d_in[7];
    const float* bqkv   = (const float*)d_in[8];
    const float* Wo     = (const float*)d_in[9];
    const float* bo     = (const float*)d_in[10];
    const float* conv_w = (const float*)d_in[11];
    const float* conv_b = (const float*)d_in[12];
    const float* gate_W = (const float*)d_in[13];
    const float* gate_b = (const float*)d_in[14];
    const float* exp_W  = (const float*)d_in[15];
    const float* exp_b  = (const float*)d_in[16];
    const float* ff_W1  = (const float*)d_in[17];
    const float* ff_b1  = (const float*)d_in[18];
    const float* ff_W2  = (const float*)d_in[19];
    const float* ff_b2  = (const float*)d_in[20];
    float* out = (float*)d_out;

    // workspace layout (bytes) -- big region is time-aliased:
    //   [qkv 48MB | attnO 16MB] -> [moe 64MB] -> [ffn hidden 64MB]
    char* w = (char*)d_ws;
    float*          xb     = (float*)(w + 0);                         // 33.5 MB
    unsigned short* hbf    = (unsigned short*)(w + 33554432);         // 16.8 MB
    unsigned short* qkv    = (unsigned short*)(w + 50331648);         // 48 MB
    unsigned short* attnO  = (unsigned short*)(w + 100663296);        // 16.8 MB
    float*          moe    = (float*)(w + 50331648);                  // alias
    unsigned short* hidden = (unsigned short*)(w + 50331648);         // alias
    float*          topkP  = (float*)(w + 117440512);
    float*          entBuf = (float*)(w + 117440512 + 65536);
    int*            ctr    = (int*)(w + 117440512 + 65536 + 32768);
    int*            idxA   = (int*)(w + 117440512 + 65536 + 32768 + 64);
    int*            idxOut = idxA + NEXP * NTOK;

    init_kernel<<<1, 32, 0, stream>>>(ctr);

    // x1 = ln1(x) ; qkv = x1 @ Wqkv + bqkv (bf16)
    ln_kernel<<<NTOK, 256, 0, stream>>>(x_in, g1, b1, hbf);
    gemm_bf16_kernel<0><<<dim3(NTOK / 128, 3 * CDIM / 128), 256, 0, stream>>>(
        hbf, CDIM, Wqkv, 3 * CDIM, 1, bqkv, nullptr, nullptr, qkv,
        NTOK, 3 * CDIM, CDIM, 0, 0, nullptr, nullptr, nullptr);

    // window attention (16 surviving windows per b,h)
    attn_kernel<<<2 * 16 * 16, 256, 0, stream>>>(qkv, attnO);

    // x = x + attnO @ Wo + bo
    gemm_bf16_kernel<0><<<dim3(NTOK / 128, CDIM / 128), 256, 0, stream>>>(
        attnO, CDIM, Wo, CDIM, 1, bo, xb, x_in, nullptr,
        NTOK, CDIM, CDIM, 0, 0, nullptr, nullptr, nullptr);

    // x = x + dconv(ln2(x)) as 3 shifted GEMMs (taps at t-2, t, t+2)
    ln_kernel<<<NTOK, 256, 0, stream>>>(xb, g2, b2, hbf);
    for (int k = 0; k < 3; ++k) {
        gemm_bf16_kernel<1><<<dim3(NTOK / 128, CDIM / 128), 256, 0, stream>>>(
            hbf, CDIM, conv_w + k, 3, 3 * CDIM,
            (k == 0) ? conv_b : nullptr, xb, xb, nullptr,
            NTOK, CDIM, CDIM, 0, (k - 1) * 2, nullptr, nullptr, nullptr);
    }

    // MoE: gate on ln3(x), compute only top-2 experts via gather/scatter
    ln_kernel<<<NTOK, 256, 0, stream>>>(xb, g3, b3, hbf);
    gate_kernel<<<NTOK / 8, 256, 0, stream>>>(hbf, gate_W, gate_b,
        topkP, entBuf, idxA, idxOut, ctr);
    aux_kernel<<<1, 256, 0, stream>>>(entBuf, ctr, out + (long long)NTOK * CDIM);
    for (int e = 0; e < NEXP; ++e) {
        gemm_bf16_kernel<2><<<dim3(NTOK / 128, CDIM / 128), 256, 0, stream>>>(
            hbf, CDIM, exp_W + (long long)e * CDIM * CDIM, CDIM, 1,
            exp_b + e * CDIM, moe, nullptr, nullptr,
            NTOK, CDIM, CDIM, 0, 0, idxA + e * NTOK, idxOut + e * NTOK, ctr + e);
    }
    combine_kernel<<<NTOK, 256, 0, stream>>>(xb, moe, topkP, hbf);

    // FFN: hidden = gelu(x @ W1 + b1) ; out = x + hidden @ W2 + b2
    gemm_bf16_kernel<0><<<dim3(NTOK / 128, FDIM / 128), 256, 0, stream>>>(
        hbf, CDIM, ff_W1, FDIM, 1, ff_b1, nullptr, nullptr, hidden,
        NTOK, FDIM, CDIM, 1, 0, nullptr, nullptr, nullptr);
    gemm_bf16_kernel<0><<<dim3(NTOK / 128, CDIM / 128), 256, 0, stream>>>(
        hidden, FDIM, ff_W2, CDIM, 1, ff_b2, out, xb, nullptr,
        NTOK, CDIM, FDIM, 0, 0, nullptr, nullptr, nullptr);
}
